// pointASNLDecoding_68650757259522
// MI455X (gfx1250) — matrix operand order repaired and checked by reading, not compiled, not run on hardware
//
#include <hip/hip_runtime.h>
#include <hip/hip_bf16.h>

typedef __attribute__((ext_vector_type(16))) _Float16 v16h;
typedef __attribute__((ext_vector_type(8)))  _Float16 v8h;
typedef __attribute__((ext_vector_type(8)))  float    v8f;

#define DEV_INLINE __device__ __forceinline__

// ---- problem constants ----
constexpr int Bc = 4, Nn = 4096, Sn = 1024;
constexpr int C1 = 128, C2 = 256, NS = 16, BCd = 64, LM = 128, M0c = 256, M1c = 256;
constexpr float BN_EPS = 1e-5f;

DEV_INLINE v16h cmb(v8h a, v8h b) {
  return __builtin_shufflevector(a, b, 0, 1, 2, 3, 4, 5, 6, 7, 8, 9, 10, 11, 12, 13, 14, 15);
}

// LDS byte offset of a shared-memory object (addrspacecast generic->AS3, then ptrtoint)
DEV_INLINE unsigned lds_addr_of(const void* p) {
  return (unsigned)(unsigned long long)(__attribute__((address_space(3))) const void*)p;
}

// Async 16B copy global -> LDS (ASYNCcnt-tracked, no VGPR round trip)
DEV_INLINE void async_copy_b128(unsigned lds_off, const void* gptr) {
  asm volatile("global_load_async_to_lds_b128 %0, %1, off"
               :
               : "v"(lds_off), "v"((unsigned long long)gptr)
               : "memory");
}
DEV_INLINE void async_wait0() {
  asm volatile("s_wait_asynccnt 0x0" ::: "memory");
}

// ======================================================================
// transpose (B,C,N) -> (B,N,C)
// ======================================================================
__global__ void transpose_kernel(const float* __restrict__ in, float* __restrict__ out,
                                 int Cn, int Npts, long total) {
  long i = (long)blockIdx.x * blockDim.x + threadIdx.x;
  if (i >= total) return;
  int n = (int)(i % Npts);
  int c = (int)((i / Npts) % Cn);
  long b = i / ((long)Npts * Cn);
  out[(b * Npts + n) * (long)Cn + c] = in[i];
}

// ======================================================================
// f32 -> f16 converters (optionally fused BN+relu); feed the WMMA pipes.
// ======================================================================
__global__ void cvt_f16(const float* __restrict__ src, _Float16* __restrict__ dst, long n) {
  long i = (long)blockIdx.x * blockDim.x + threadIdx.x;
  if (i < n) dst[i] = (_Float16)src[i];
}

__global__ void cvt_bnrelu_f16(const float* __restrict__ src, const float* __restrict__ scale,
                               const float* __restrict__ shift, int C,
                               _Float16* __restrict__ dst, long n) {
  long i = (long)blockIdx.x * blockDim.x + threadIdx.x;
  if (i >= n) return;
  int c = (int)(i % C);
  dst[i] = (_Float16)fmaxf(src[i] * scale[c] + shift[c], 0.f);
}

// A for c0: [relu(bn(pc_pre)) | p1] per row, as f16
__global__ void concat_c0_f16(const float* __restrict__ pc_pre, const float* __restrict__ p1t,
                              const float* __restrict__ scale, const float* __restrict__ shift,
                              _Float16* __restrict__ dst, long total) {
  long i = (long)blockIdx.x * blockDim.x + threadIdx.x;
  if (i >= total) return;
  long row = i >> 8;
  int c = (int)(i & 255);
  float v;
  if (c < 128) v = fmaxf(pc_pre[row * 128 + c] * scale[c] + shift[c], 0.f);
  else         v = p1t[row * 128 + (c - 128)];
  dst[i] = (_Float16)v;
}

// ======================================================================
// kNN: per query thread, top-16 smallest squared distances.
// ======================================================================
__global__ void knn_kernel(const float* __restrict__ q, const float* __restrict__ cand,
                           int nq, int ncand, int* __restrict__ idx, float* __restrict__ wgt) {
  __shared__ float sx[1024], sy[1024], sz[1024];
  int b = blockIdx.y;
  int n = blockIdx.x * blockDim.x + threadIdx.x;
  const float* qp = q + ((long)b * nq + n) * 3;
  float qx = qp[0], qy = qp[1], qz = qp[2];
  float bd[16]; int bi[16];
#pragma unroll
  for (int u = 0; u < 16; u++) { bd[u] = 3.4e38f; bi[u] = 0; }
  float worst = 3.4e38f;
  for (int j0 = 0; j0 < ncand; j0 += 1024) {
    int chunk = min(1024, ncand - j0);
    __syncthreads();
    for (int t = threadIdx.x; t < chunk; t += blockDim.x) {
      const float* cp = cand + ((long)b * ncand + j0 + t) * 3;
      sx[t] = cp[0]; sy[t] = cp[1]; sz[t] = cp[2];
    }
    __syncthreads();
    for (int t = 0; t < chunk; t++) {
      float dx = qx - sx[t], dy = qy - sy[t], dz = qz - sz[t];
      float d = dx * dx + dy * dy + dz * dz;
      if (d < worst) {
        bool rep = false;
#pragma unroll
        for (int u = 0; u < 16; u++)
          if (!rep && bd[u] == worst) { bd[u] = d; bi[u] = j0 + t; rep = true; }
        worst = bd[0];
#pragma unroll
        for (int u = 1; u < 16; u++) worst = fmaxf(worst, bd[u]);
      }
    }
  }
  long base = ((long)b * nq + n) * 16;
#pragma unroll
  for (int u = 0; u < 16; u++) idx[base + u] = bi[u];
  if (wgt) {
    float rc[16]; float rsum = 0.f;
#pragma unroll
    for (int u = 0; u < 16; u++) { rc[u] = 1.0f / (bd[u] + 1e-8f); rsum += rc[u]; }
    float inv = 1.0f / rsum;
#pragma unroll
    for (int u = 0; u < 16; u++) wgt[base + u] = rc[u] * inv;
  }
}

// ======================================================================
// WMMA GEMM:  C[rows,cols] = A[rows,K] * W[cols,K]^T, f16 operands, f32 acc.
// Per-lane fragments are contiguous 16B chunks in row-major f16 memory:
//   A lane(row,hi): k in [kb+8hi, +8) and [kb+16+8hi, +8)   -> 2x b128
//   B lane(col,hi): k in [kb+8hi, +8) and [kb+16+8hi, +8)   -> 2x b128
// One wave per 16x16 tile, 4 waves/block.
// ======================================================================
__global__ void gemm_f16(const _Float16* __restrict__ A, const _Float16* __restrict__ W,
                         float* __restrict__ C, int rows, int K, int cols) {
  int wave = threadIdx.x >> 5;
  int lane = threadIdx.x & 31;
  int rt = blockIdx.x;
  int ct = blockIdx.y * 4 + wave;
  int colsT = cols >> 4;
  if (ct >= colsT) return;
  int lo = lane & 15, hi = lane >> 4;
  const _Float16* Ar = A + (long)(rt * 16 + lo) * K;
  const _Float16* Wr = W + (long)(ct * 16 + lo) * K;
  v8f acc = {};
  for (int kb = 0; kb < K; kb += 32) {
    v8h a0 = *(const v8h*)(Ar + kb + 8 * hi);
    v8h a1 = *(const v8h*)(Ar + kb + 16 + 8 * hi);
    v8h b0 = *(const v8h*)(Wr + kb + 8 * hi);
    v8h b1 = *(const v8h*)(Wr + kb + 16 + 8 * hi);
    acc = __builtin_amdgcn_wmma_f32_16x16x32_f16(false, cmb(a0, a1), false, cmb(b0, b1),
                                                 (short)0, acc, false, false);
  }
#pragma unroll
  for (int r = 0; r < 8; r++) {
    int gr = rt * 16 + r + 8 * hi;
    int gc = ct * 16 + lo;
    C[(long)gr * cols + gc] = acc[r];
  }
}

// ======================================================================
// Online-softmax attention: one wave per query row (d=64 -> 2 comps/lane)
// ======================================================================
__global__ void attn_kernel(const float* __restrict__ nlq, const float* __restrict__ nlkv,
                            float* __restrict__ nlp0) {
  int gw = (int)((blockIdx.x * (long)blockDim.x + threadIdx.x) >> 5);
  int lane = threadIdx.x & 31;
  int b = gw / Sn, s = gw % Sn;
  const float* qp = nlq + ((long)b * Sn + s) * 64;
  float q0 = qp[2 * lane], q1 = qp[2 * lane + 1];
  float m = -3.4e38f, l = 0.f, a0 = 0.f, a1 = 0.f;
  const float* kvb = nlkv + (long)b * Nn * 128;
  for (int n = 0; n < Nn; n++) {
    const float* kp = kvb + (long)n * 128;
    float p = q0 * kp[2 * lane] + q1 * kp[2 * lane + 1];
#pragma unroll
    for (int off = 16; off; off >>= 1) p += __shfl_xor(p, off, 32);
    float sv = p * 0.125f;                       // 1/sqrt(64)
    float mn = fmaxf(m, sv);
    float corr = __expf(m - mn);
    float e = __expf(sv - mn);
    l = l * corr + e;
    a0 = a0 * corr + e * kp[64 + 2 * lane];
    a1 = a1 * corr + e * kp[64 + 2 * lane + 1];
    m = mn;
  }
  float* op = nlp0 + ((long)b * Sn + s) * 64;
  float inv = 1.0f / l;
  op[2 * lane] = a0 * inv;
  op[2 * lane + 1] = a1 * inv;
}

// ======================================================================
// BN stats over rows (one block per channel). Channel biases cancel in BN.
// ======================================================================
__global__ void stats_kernel(const float* __restrict__ X, int rows, int C,
                             const float* __restrict__ g, const float* __restrict__ be,
                             float* __restrict__ scale, float* __restrict__ shift) {
  __shared__ float ss[256], sq[256];
  int c = blockIdx.x, tid = threadIdx.x;
  float s = 0.f, q = 0.f;
  for (int r = tid; r < rows; r += blockDim.x) {
    float v = X[(long)r * C + c];
    s += v; q += v * v;
  }
  ss[tid] = s; sq[tid] = q; __syncthreads();
  for (int off = 128; off; off >>= 1) {
    if (tid < off) { ss[tid] += ss[tid + off]; sq[tid] += sq[tid + off]; }
    __syncthreads();
  }
  if (tid == 0) {
    float mean = ss[0] / rows;
    float var = fmaxf(sq[0] / rows - mean * mean, 0.f);
    float sc = g[c] * rsqrtf(var + BN_EPS);
    scale[c] = sc;
    shift[c] = be[c] - mean * sc;
  }
}

__global__ void finalize_stats(const float* __restrict__ sum, const float* __restrict__ sq,
                               const float* __restrict__ g, const float* __restrict__ be,
                               float* __restrict__ scale, float* __restrict__ shift,
                               int C, float inv_cnt) {
  int c = blockIdx.x * blockDim.x + threadIdx.x;
  if (c >= C) return;
  float mean = sum[c] * inv_cnt;
  float var = fmaxf(sq[c] * inv_cnt - mean * mean, 0.f);
  float sc = g[c] * rsqrtf(var + BN_EPS);
  scale[c] = sc;
  shift[c] = be[c] - mean * sc;
}

__global__ void zero_kernel(float* __restrict__ p, int n) {
  int i = blockIdx.x * blockDim.x + threadIdx.x;
  if (i < n) p[i] = 0.f;
}

// p2 += relu(bn(nlp_pre))
__global__ void p2_update(float* __restrict__ p2t, const float* __restrict__ pre,
                          const float* __restrict__ scale, const float* __restrict__ shift,
                          long total) {
  long i = (long)blockIdx.x * blockDim.x + threadIdx.x;
  if (i >= total) return;
  int c = (int)(i & 255);
  p2t[i] += fmaxf(pre[i] * scale[c] + shift[c], 0.f);
}

// interp[b,n,c] = sum_k w[b,n,k] * p2[b, idx[b,n,k], c]
__global__ void interp_kernel(const float* __restrict__ p2t, const int* __restrict__ idx,
                              const float* __restrict__ wgt, float* __restrict__ interp) {
  __shared__ int si[16];
  __shared__ float sw[16];
  long bn = blockIdx.x;
  long b = bn / Nn;
  if (threadIdx.x < 16) {
    si[threadIdx.x] = idx[bn * 16 + threadIdx.x];
    sw[threadIdx.x] = wgt[bn * 16 + threadIdx.x];
  }
  __syncthreads();
  int c = threadIdx.x;
  float acc = 0.f;
#pragma unroll
  for (int k = 0; k < 16; k++) acc += sw[k] * p2t[((long)b * Sn + si[k]) * C2 + c];
  interp[bn * C2 + c] = acc;
}

// ======================================================================
// Per-point fused kernel (block = 128 thr / 4 waves per point):
// async-gather gfeat(f16) into LDS (ASYNCcnt) overlapped with the
// independent wf stage, then z0 = gfeat @ w_l0^T (WMMA, LDS b128
// fragments); pass1: BN stats, pass2: BN+relu -> m -> pc.
// ======================================================================
template <bool STATS>
__global__ void ppoint_kernel(const _Float16* __restrict__ interph, const int* __restrict__ gidx,
                              const float* __restrict__ x1t,
                              const _Float16* __restrict__ w_l0h, const float* __restrict__ w_wn,
                              const float* __restrict__ w_pc,
                              float* __restrict__ l0_sum, float* __restrict__ l0_sq,
                              float* __restrict__ wn_sum, float* __restrict__ wn_sq,
                              const float* __restrict__ l0_scale, const float* __restrict__ l0_shift,
                              const float* __restrict__ wn_scale, const float* __restrict__ wn_shift,
                              float* __restrict__ pc_pre) {
  __shared__ int sh_idx[16];
  __shared__ float sh_ctr[3];
  __shared__ __align__(16) _Float16 sh_gf[16][C2 + 8];   // row stride 528B (16B multiple)
  __shared__ _Float16 sh_np[16][LM + 8];
  __shared__ float sh_zw[16][33];
  __shared__ float sh_m[128][33];
  long bn = blockIdx.x;
  int b = (int)(bn / Nn), n = (int)(bn % Nn);
  int tid = threadIdx.x;
  if (tid < 16) sh_idx[tid] = gidx[bn * 16 + tid];
  if (tid >= 16 && tid < 19) sh_ctr[tid - 16] = x1t[((long)b * Nn + n) * 3 + (tid - 16)];
  __syncthreads();

  // ---- issue async gather of neighbor features: 16B per lane per op ----
  for (int e = tid; e < 16 * (C2 / 8); e += 128) {
    int k = e >> 5, c8 = e & 31;
    async_copy_b128(lds_addr_of(&sh_gf[k][c8 * 8]),
                    &interph[((long)b * Nn + sh_idx[k]) * C2 + c8 * 8]);
  }

  // ---- overlap: wf stage only needs sh_idx/x1t (independent of sh_gf) ----
  for (int e = tid; e < 16 * 32; e += 128) {
    int k = e >> 5, c = e & 31;
    const float* pg = x1t + ((long)b * Nn + sh_idx[k]) * 3;
    float z = w_wn[c * 3 + 0] * (pg[0] - sh_ctr[0]) +
              w_wn[c * 3 + 1] * (pg[1] - sh_ctr[1]) +
              w_wn[c * 3 + 2] * (pg[2] - sh_ctr[2]);
    sh_zw[k][c] = STATS ? z : fmaxf(z * wn_scale[c] + wn_shift[c], 0.f);
  }

  async_wait0();       // per-wave ASYNCcnt drain, then cross-wave barrier
  __syncthreads();

  int wave = tid >> 5, lane = tid & 31;
  int lo = lane & 15, hi = lane >> 4;
  for (int ct = wave; ct < LM / 16; ct += 4) {     // 8 col tiles of z0 (16x128)
    v8f acc = {};
    const _Float16* Wr = w_l0h + (long)(ct * 16 + lo) * C2;
    for (int ks = 0; ks < C2 / 32; ks++) {
      int kb = ks * 32;
      v8h a0 = *(const v8h*)&sh_gf[lo][kb + 8 * hi];
      v8h a1 = *(const v8h*)&sh_gf[lo][kb + 16 + 8 * hi];
      v8h b0 = *(const v8h*)(Wr + kb + 8 * hi);
      v8h b1 = *(const v8h*)(Wr + kb + 16 + 8 * hi);
      acc = __builtin_amdgcn_wmma_f32_16x16x32_f16(false, cmb(a0, a1), false, cmb(b0, b1),
                                                   (short)0, acc, false, false);
    }
    if (STATS) {
      float s = 0.f, q = 0.f;
#pragma unroll
      for (int r = 0; r < 8; r++) { s += acc[r]; q += acc[r] * acc[r]; }
      s += __shfl_xor(s, 16, 32);
      q += __shfl_xor(q, 16, 32);
      if (hi == 0) {
        int ch = ct * 16 + lo;
        atomicAdd(&l0_sum[ch], s);
        atomicAdd(&l0_sq[ch], q);
      }
    } else {
      int ch = ct * 16 + lo;
      float sc = l0_scale[ch], sh = l0_shift[ch];
#pragma unroll
      for (int r = 0; r < 8; r++) {
        int krow = r + 8 * hi;
        sh_np[krow][ch] = (_Float16)fmaxf(acc[r] * sc + sh, 0.f);
      }
    }
  }
  __syncthreads();
  if (STATS) {
    if (tid < 32) {
      float s = 0.f, q = 0.f;
#pragma unroll
      for (int k = 0; k < 16; k++) { float v = sh_zw[k][tid]; s += v; q += v * v; }
      atomicAdd(&wn_sum[tid], s);
      atomicAdd(&wn_sq[tid], q);
    }
    return;
  }
  // m[l][w] = sum_k npts[k][l] * wf[k][w]
  {
    float mreg[32];
#pragma unroll
    for (int w = 0; w < 32; w++) mreg[w] = 0.f;
    int l = tid;
#pragma unroll
    for (int k = 0; k < 16; k++) {
      float nv = (float)sh_np[k][l];
#pragma unroll
      for (int w = 0; w < 32; w++) mreg[w] += nv * sh_zw[k][w];
    }
#pragma unroll
    for (int w = 0; w < 32; w++) sh_m[l][w] = mreg[w];
  }
  __syncthreads();
  // pc[o] = sum_{w,l} m[l][w] * w_pc[o,w,l]   (w_pc is L2-resident, prefetch ahead)
  {
    int o = tid;
    float acc = 0.f;
    for (int w = 0; w < 32; w++) {
      const float* wp = w_pc + (long)o * 4096 + w * 128;
      if (w + 1 < 32) __builtin_prefetch(wp + 128, 0, 3);
#pragma unroll 8
      for (int l2 = 0; l2 < 128; l2++) acc += sh_m[l2][w] * wp[l2];
    }
    pc_pre[bn * LM + o] = acc;
  }
}

// out[b,o,n] = relu(bn(feat1[b,n,o]))
__global__ void out_kernel(const float* __restrict__ feat1, const float* __restrict__ scale,
                           const float* __restrict__ shift, float* __restrict__ out) {
  long i = (long)blockIdx.x * blockDim.x + threadIdx.x;
  long total = (long)Bc * M1c * Nn;
  if (i >= total) return;
  int n = (int)(i % Nn);
  int o = (int)((i / Nn) % M1c);
  long b = i / ((long)Nn * M1c);
  float v = feat1[((long)b * Nn + n) * M1c + o];
  out[i] = fmaxf(v * scale[o] + shift[o], 0.f);
}

// ======================================================================
extern "C" void kernel_launch(void* const* d_in, const int* in_sizes, int n_in,
                              void* d_out, int out_size, void* d_ws, size_t ws_size,
                              hipStream_t stream) {
  (void)in_sizes; (void)n_in; (void)out_size; (void)ws_size;
  const float* xyz1    = (const float*)d_in[0];
  const float* xyz2    = (const float*)d_in[1];
  const float* points1 = (const float*)d_in[2];
  const float* points2 = (const float*)d_in[3];
  const float* w_nlq   = (const float*)d_in[4];
  const float* w_nlkv  = (const float*)d_in[5];
  const float* w_nl    = (const float*)d_in[6];
  const float* g_nl    = (const float*)d_in[7];
  const float* be_nl   = (const float*)d_in[8];
  const float* w_wn    = (const float*)d_in[9];
  const float* g_wn    = (const float*)d_in[11];
  const float* be_wn   = (const float*)d_in[12];
  const float* w_l0    = (const float*)d_in[13];
  const float* g_l0    = (const float*)d_in[15];
  const float* be_l0   = (const float*)d_in[16];
  const float* w_pc    = (const float*)d_in[17];
  const float* g_pc    = (const float*)d_in[19];
  const float* be_pc   = (const float*)d_in[20];
  const float* w_c0    = (const float*)d_in[21];
  const float* g_c0    = (const float*)d_in[23];
  const float* be_c0   = (const float*)d_in[24];
  const float* w_c1    = (const float*)d_in[25];
  const float* g_c1    = (const float*)d_in[27];
  const float* be_c1   = (const float*)d_in[28];
  float* out = (float*)d_out;

  float* ws = (float*)d_ws;
  size_t off = 0;
  auto alloc  = [&](size_t nf) { float* p = ws + off; off += nf; return p; };
  auto allocH = [&](size_t nh) { _Float16* p = (_Float16*)(ws + off); off += (nh + 1) / 2; return p; };
  // f32 buffers
  float* x1t     = alloc((size_t)Bc * Nn * 3);
  float* x2t     = alloc((size_t)Bc * Sn * 3);
  float* p1t     = alloc((size_t)Bc * Nn * C1);
  float* p2t     = alloc((size_t)Bc * Sn * C2);
  int*   idx     = (int*)alloc((size_t)Bc * Nn * NS);
  float* wgt     = alloc((size_t)Bc * Nn * NS);
  int*   gidx    = (int*)alloc((size_t)Bc * Nn * NS);
  float* nlq     = alloc((size_t)Bc * Sn * BCd);
  float* nlkv    = alloc((size_t)Bc * Nn * 128);
  float* nlp0    = alloc((size_t)Bc * Sn * BCd);
  float* nlp_pre = alloc((size_t)Bc * Sn * C2);
  float* interp  = alloc((size_t)Bc * Nn * C2);
  float* pc_pre  = alloc((size_t)Bc * Nn * LM);
  float* feat0   = alloc((size_t)Bc * Nn * M0c);
  float* feat1   = alloc((size_t)Bc * Nn * M1c);
  // f16 operand buffers
  _Float16* p2h     = allocH((size_t)Bc * Sn * C2);
  _Float16* p1h     = allocH((size_t)Bc * Nn * C1);
  _Float16* nlp0h   = allocH((size_t)Bc * Sn * BCd);
  _Float16* interph = allocH((size_t)Bc * Nn * C2);
  _Float16* Ac0h    = allocH((size_t)Bc * Nn * (LM + C1));
  _Float16* feat0h  = allocH((size_t)Bc * Nn * M0c);
  _Float16* w_nlqh  = allocH((size_t)BCd * C2);
  _Float16* w_nlkvh = allocH((size_t)128 * C1);
  _Float16* w_nlh   = allocH((size_t)C2 * BCd);
  _Float16* w_l0h   = allocH((size_t)LM * C2);
  _Float16* w_c0h   = allocH((size_t)M0c * (LM + C1));
  _Float16* w_c1h   = allocH((size_t)M1c * M0c);
  // stats
  float* snl_scale = alloc(256); float* snl_shift = alloc(256);
  float* atom = alloc(320);   // l0_sum(128) l0_sq(128) wn_sum(32) wn_sq(32)
  float* l0_sum = atom, *l0_sq = atom + 128, *wn_sum = atom + 256, *wn_sq = atom + 288;
  float* l0_scale = alloc(128); float* l0_shift = alloc(128);
  float* wn_scale = alloc(32);  float* wn_shift = alloc(32);
  float* spc_scale = alloc(128); float* spc_shift = alloc(128);
  float* sc0_scale = alloc(256); float* sc0_shift = alloc(256);
  float* sc1_scale = alloc(256); float* sc1_shift = alloc(256);

  auto CVT = [&](const float* s, _Float16* d, long n) {
    cvt_f16<<<(n + 255) / 256, 256, 0, stream>>>(s, d, n);
  };

  // 0) zero atomic accumulators
  zero_kernel<<<2, 256, 0, stream>>>(atom, 320);

  // 1) transposes (B,C,N)->(B,N,C)
  {
    long t;
    t = (long)Bc * 3 * Nn;
    transpose_kernel<<<(t + 255) / 256, 256, 0, stream>>>(xyz1, x1t, 3, Nn, t);
    t = (long)Bc * 3 * Sn;
    transpose_kernel<<<(t + 255) / 256, 256, 0, stream>>>(xyz2, x2t, 3, Sn, t);
    t = (long)Bc * C1 * Nn;
    transpose_kernel<<<(t + 255) / 256, 256, 0, stream>>>(points1, p1t, C1, Nn, t);
    t = (long)Bc * C2 * Sn;
    transpose_kernel<<<(t + 255) / 256, 256, 0, stream>>>(points2, p2t, C2, Sn, t);
  }

  // 2) f16 operand conversion (weights + pre-update activations)
  CVT(p2t, p2h, (long)Bc * Sn * C2);
  CVT(p1t, p1h, (long)Bc * Nn * C1);
  CVT(w_nlq, w_nlqh, (long)BCd * C2);
  CVT(w_nlkv, w_nlkvh, (long)128 * C1);
  CVT(w_nl, w_nlh, (long)C2 * BCd);
  CVT(w_l0, w_l0h, (long)LM * C2);
  CVT(w_c0, w_c0h, (long)M0c * (LM + C1));
  CVT(w_c1, w_c1h, (long)M1c * M0c);

  // 3) kNN: x1 vs x2 (idx + weights), x1 vs x1 (gidx)
  knn_kernel<<<dim3(Nn / 256, Bc), 256, 0, stream>>>(x1t, x2t, Nn, Sn, idx, wgt);
  knn_kernel<<<dim3(Nn / 256, Bc), 256, 0, stream>>>(x1t, x1t, Nn, Nn, gidx, nullptr);

  // 4) nlq = p2 @ w_nlq^T ; nlkv = p1 @ w_nlkv^T
  gemm_f16<<<dim3(Bc * Sn / 16, 1), 128, 0, stream>>>(p2h, w_nlqh, nlq, Bc * Sn, C2, BCd);
  gemm_f16<<<dim3(Bc * Nn / 16, 2), 128, 0, stream>>>(p1h, w_nlkvh, nlkv, Bc * Nn, C1, 128);
  // 5) attention (flash-style, wave per query)
  attn_kernel<<<(Bc * Sn) / 8, 256, 0, stream>>>(nlq, nlkv, nlp0);
  // 6) nlp_pre = nlp0 @ w_nl^T
  CVT(nlp0, nlp0h, (long)Bc * Sn * BCd);
  gemm_f16<<<dim3(Bc * Sn / 16, 4), 128, 0, stream>>>(nlp0h, w_nlh, nlp_pre, Bc * Sn, BCd, C2);
  // 7) BN stats + p2 += relu(bn(nlp_pre))
  stats_kernel<<<C2, 256, 0, stream>>>(nlp_pre, Bc * Sn, C2, g_nl, be_nl, snl_scale, snl_shift);
  p2_update<<<(Bc * Sn * C2) / 256, 256, 0, stream>>>(p2t, nlp_pre, snl_scale, snl_shift,
                                                     (long)Bc * Sn * C2);
  // 8) interpolation (then f16 copy for the gathered WMMA stage)
  interp_kernel<<<Bc * Nn, 256, 0, stream>>>(p2t, idx, wgt, interp);
  CVT(interp, interph, (long)Bc * Nn * C2);

  // 9) per-point pass 1 (BN stats), finalize, pass 2 (-> pc_pre)
  ppoint_kernel<true><<<Bc * Nn, 128, 0, stream>>>(
      interph, gidx, x1t, w_l0h, w_wn, w_pc, l0_sum, l0_sq, wn_sum, wn_sq,
      nullptr, nullptr, nullptr, nullptr, nullptr);
  {
    float inv_cnt = 1.0f / ((float)Bc * Nn * NS);
    finalize_stats<<<1, 128, 0, stream>>>(l0_sum, l0_sq, g_l0, be_l0, l0_scale, l0_shift,
                                          128, inv_cnt);
    finalize_stats<<<1, 32, 0, stream>>>(wn_sum, wn_sq, g_wn, be_wn, wn_scale, wn_shift,
                                         32, inv_cnt);
  }
  ppoint_kernel<false><<<Bc * Nn, 128, 0, stream>>>(
      interph, gidx, x1t, w_l0h, w_wn, w_pc, nullptr, nullptr, nullptr, nullptr,
      l0_scale, l0_shift, wn_scale, wn_shift, pc_pre);

  // 10) pc BN stats; assemble f16 A = [bn(pc)|p1]; c0 GEMM
  stats_kernel<<<LM, 256, 0, stream>>>(pc_pre, Bc * Nn, LM, g_pc, be_pc, spc_scale, spc_shift);
  concat_c0_f16<<<((long)Bc * Nn * 256) / 256, 256, 0, stream>>>(pc_pre, p1t, spc_scale,
                                                                 spc_shift, Ac0h,
                                                                 (long)Bc * Nn * 256);
  gemm_f16<<<dim3(Bc * Nn / 16, 4), 128, 0, stream>>>(Ac0h, w_c0h, feat0, Bc * Nn,
                                                      LM + C1, M0c);
  // 11) c0 BN stats; fused bn+relu f16 conversion; c1 GEMM
  stats_kernel<<<M0c, 256, 0, stream>>>(feat0, Bc * Nn, M0c, g_c0, be_c0, sc0_scale, sc0_shift);
  cvt_bnrelu_f16<<<((long)Bc * Nn * M0c) / 256, 256, 0, stream>>>(
      feat0, sc0_scale, sc0_shift, M0c, feat0h, (long)Bc * Nn * M0c);
  gemm_f16<<<dim3(Bc * Nn / 16, 4), 128, 0, stream>>>(feat0h, w_c1h, feat1, Bc * Nn, M0c, M1c);
  // 12) c1 BN stats; final relu(bn(.)) + transpose to (B, 256, N)
  stats_kernel<<<M1c, 256, 0, stream>>>(feat1, Bc * Nn, M1c, g_c1, be_c1, sc1_scale, sc1_shift);
  out_kernel<<<((long)Bc * M1c * Nn) / 256, 256, 0, stream>>>(feat1, sc1_scale, sc1_shift, out);
}
